// Vanilla_RNN_9500467659081
// MI455X (gfx1250) — compile-verified, matrix operand order checked by reading, not run on hardware
//
#include <hip/hip_runtime.h>
#include <hip/hip_bf16.h>

// ---------------------------------------------------------------------------
// Vanilla RNN on MI455X (gfx1250):
//   xi     = emb[x] @ W_ih^T + b_ih               (bf16 WMMA)
//   h_t    = tanh(xi_t + h_{t-1} @ W_hh^T + b_hh) (single-WGP LDS recurrence, WMMA)
//   logits = hs @ W_out^T + b_out                 (bf16 WMMA + TDM-staged W_out tiles)
// ---------------------------------------------------------------------------

#define VOCAB 10000
#define EMB   512
#define HID   1024
#define BB    16
#define TT    512
#define MROWS (BB * TT)   // 8192

typedef __attribute__((ext_vector_type(8)))  __bf16 bf16x8;
typedef __attribute__((ext_vector_type(16))) __bf16 bf16x16;
typedef __attribute__((ext_vector_type(8)))  float  f32x8;
typedef __attribute__((ext_vector_type(4)))  unsigned int u32x4;
typedef __attribute__((ext_vector_type(8)))  int    i32x8;
typedef __attribute__((ext_vector_type(4)))  int    i32x4;

// ---- fragment loader: 16 bf16 along K for one row (A) or one W-row (B=W^T col)
// lane<16 : K in [kk, kk+8) and [kk+16, kk+24)
// lane>=16: K in [kk+8, kk+16) and [kk+24, kk+32)
__device__ __forceinline__ bf16x16 load_frag_row(const __bf16* rowPtr, int kk, int lane) {
  int off = kk + ((lane & 16) ? 8 : 0);
  bf16x8 lo = *(const bf16x8*)(rowPtr + off);
  bf16x8 hi = *(const bf16x8*)(rowPtr + off + 16);
  bf16x16 r;
#pragma unroll
  for (int i = 0; i < 8; ++i) { r[i] = lo[i]; r[i + 8] = hi[i]; }
  return r;
}

__device__ __forceinline__ f32x8 wmma_bf16(bf16x16 a, bf16x16 b, f32x8 c) {
  return __builtin_amdgcn_wmma_f32_16x16x32_bf16(
      /*neg_a=*/false, a, /*neg_b=*/false, b,
      /*c_mod=*/(short)0, c, /*reuse_a=*/false, /*reuse_b=*/false);
}

// ---------------------------------------------------------------------------
// Tensor Data Mover: 2D tile (tile_k x tile_rows, bf16) from row-major global
// matrix (row_stride elems) into LDS at lds_off. D# per CDNA5 ISA ch.8.
// This toolchain exposes the 6-arg builtin:
//   (uint32x4 g0, int32x8 g1, int32x4 g2, int32x4 g3, int32x8 g4, i32 cpol)
// ---------------------------------------------------------------------------
__device__ __forceinline__ void tdm_load_2d(unsigned lds_off, const __bf16* gptr,
                                            unsigned tile_k, unsigned tile_rows,
                                            unsigned row_stride,
                                            unsigned tensor_d0, unsigned tensor_d1) {
  unsigned long long ga = (unsigned long long)(size_t)gptr;

  u32x4 g0;
  g0[0] = 1u;                                              // count=1, user desc
  g0[1] = lds_off;                                         // lds_addr (bytes)
  g0[2] = (unsigned)(ga & 0xFFFFFFFFu);                    // global_addr[31:0]
  g0[3] = (unsigned)((ga >> 32) & 0x01FFFFFFu) | (2u << 30); // addr[56:32] | type=2

  i32x8 g1;
  g1[0] = (int)(1u << 16);                                 // data_size=1 (2B); mask=0
  g1[1] = (int)((tensor_d0 & 0xFFFFu) << 16);              // abar=0 | tensor_dim0 lo16
  g1[2] = (int)((tensor_d0 >> 16) | ((tensor_d1 & 0xFFFFu) << 16));
  g1[3] = (int)((tensor_d1 >> 16) | (tile_k << 16));       // tensor_dim1 hi | tile_dim0
  g1[4] = (int)(tile_rows & 0xFFFFu);                      // tile_dim1 | tile_dim2=0
  g1[5] = (int)row_stride;                                 // tensor_dim0_stride[31:0]
  g1[6] = 0;                                               // stride hi | dim1_stride lo
  g1[7] = 0;

  i32x4 z4 = {0, 0, 0, 0};
  i32x8 z8 = {0, 0, 0, 0, 0, 0, 0, 0};
  __builtin_amdgcn_tensor_load_to_lds(g0, g1, z4, z4, z8, /*cpol=*/0);
}

// ---------------------------------------------------------------------------
// Kernel 0: f32 -> bf16 conversion (grid-stride)
// ---------------------------------------------------------------------------
__global__ void k_f32_to_bf16(const float* __restrict__ src,
                              __bf16* __restrict__ dst, int n) {
  for (int i = blockIdx.x * blockDim.x + threadIdx.x; i < n;
       i += gridDim.x * blockDim.x) {
    dst[i] = (__bf16)src[i];
  }
}

// ---------------------------------------------------------------------------
// Kernel 1: xi[r, h] = emb[x[r], :] @ W_ih[h, :] + b_ih[h]   (r = b*T + t)
// Wave tile: 16 rows x 64 cols. Block: 8 waves along M. grid = (64, 16)
// ---------------------------------------------------------------------------
__global__ void k_xi_gemm(const int*    __restrict__ x,      // [B*T]
                          const __bf16* __restrict__ embB,   // [VOCAB, EMB]
                          const __bf16* __restrict__ WihB,   // [HID, EMB]
                          const float*  __restrict__ b_ih,   // [HID]
                          float*        __restrict__ xiF) {  // [B*T, HID]
  const int lane = threadIdx.x & 31;
  const int wid  = threadIdx.x >> 5;
  const int row0 = (blockIdx.x * 8 + wid) * 16;
  const int n0   = blockIdx.y * 64;

  const int   arow  = row0 + (lane & 15);
  const int   token = x[arow];
  const __bf16* aPtr = embB + (size_t)token * EMB;

  f32x8 acc[4];
  f32x8 z = {};
#pragma unroll
  for (int j = 0; j < 4; ++j) acc[j] = z;

  for (int kk = 0; kk < EMB; kk += 32) {
    bf16x16 a = load_frag_row(aPtr, kk, lane);
#pragma unroll
    for (int j = 0; j < 4; ++j) {
      int n = n0 + j * 16 + (lane & 15);
      bf16x16 b = load_frag_row(WihB + (size_t)n * EMB, kk, lane);
      acc[j] = wmma_bf16(a, b, acc[j]);
    }
  }

  const int mAdd = (lane & 16) ? 8 : 0;
#pragma unroll
  for (int j = 0; j < 4; ++j) {
    int col = n0 + j * 16 + (lane & 15);
    float bias = b_ih[col];
#pragma unroll
    for (int r = 0; r < 8; ++r) {
      int row = row0 + r + mAdd;
      xiF[(size_t)row * HID + col] = acc[j][r] + bias;
    }
  }
}

// ---------------------------------------------------------------------------
// Kernel 2: sequential recurrence, one workgroup (32 waves), h in LDS (bf16,
// double buffered). Each wave owns 2 column tiles (32 cols) of h_next.
// ---------------------------------------------------------------------------
__global__ void __launch_bounds__(1024)
k_recurrence(const float*  __restrict__ xiF,    // [B*T, HID]
             const float*  __restrict__ h0,     // [B, HID]
             const __bf16* __restrict__ WhhB,   // [HID, HID]
             const float*  __restrict__ b_hh,   // [HID]
             __bf16*       __restrict__ hsB,    // [B*T, HID] bf16
             float*        __restrict__ hT) {   // [B, HID]
  __shared__ __align__(16) __bf16 hbuf[2][BB][HID];   // 64 KB of 320 KB

  const int tid  = threadIdx.x;
  const int lane = tid & 31;
  const int wid  = tid >> 5;

  for (int i = tid; i < BB * HID; i += 1024) {
    hbuf[0][i / HID][i % HID] = (__bf16)h0[i];
  }
  __syncthreads();

  const int mAdd = (lane & 16) ? 8 : 0;
  const __bf16* aRow0 = &hbuf[0][lane & 15][0];
  const __bf16* aRow1 = &hbuf[1][lane & 15][0];

  for (int t = 0; t < TT; ++t) {
    const int cur = t & 1;
    const __bf16* aPtr = cur ? aRow1 : aRow0;

    f32x8 acc0 = {};
    f32x8 acc1 = {};
    const int j0  = wid * 2;
    const int n0a = j0 * 16 + (lane & 15);
    const int n0b = (j0 + 1) * 16 + (lane & 15);
    const __bf16* wRow0 = WhhB + (size_t)n0a * HID;
    const __bf16* wRow1 = WhhB + (size_t)n0b * HID;

    for (int kk = 0; kk < HID; kk += 32) {
      bf16x16 a  = load_frag_row(aPtr,  kk, lane);
      bf16x16 b0 = load_frag_row(wRow0, kk, lane);
      bf16x16 b1 = load_frag_row(wRow1, kk, lane);
      acc0 = wmma_bf16(a, b0, acc0);
      acc1 = wmma_bf16(a, b1, acc1);
    }

    const float bias0 = b_hh[n0a];
    const float bias1 = b_hh[n0b];
#pragma unroll
    for (int r = 0; r < 8; ++r) {
      const int m = r + mAdd;                 // batch index
      const size_t rowOff = (size_t)(m * TT + t) * HID;
      float v0 = tanhf(acc0[r] + xiF[rowOff + n0a] + bias0);
      float v1 = tanhf(acc1[r] + xiF[rowOff + n0b] + bias1);
      __bf16 bv0 = (__bf16)v0, bv1 = (__bf16)v1;
      hbuf[cur ^ 1][m][n0a] = bv0;
      hbuf[cur ^ 1][m][n0b] = bv1;
      hsB[rowOff + n0a] = bv0;
      hsB[rowOff + n0b] = bv1;
      if (t == TT - 1) {
        hT[(size_t)m * HID + n0a] = v0;
        hT[(size_t)m * HID + n0b] = v1;
      }
    }
    __syncthreads();
  }
}

// ---------------------------------------------------------------------------
// Kernel 3: logits[r, v] = hs[r, :] @ W_out[v, :] + b_out[v]
// Block tile: 128 rows x 80 cols, 8 waves along M. The 80x128 W_out K-chunk is
// shared by all 8 waves -> staged into LDS by the Tensor Data Mover, double
// buffered (issue chunk c+1 while computing chunk c). grid = (125, 64)
// ---------------------------------------------------------------------------
#define KC 128                      // K-chunk (4 wmma k-steps)
#define NKC (HID / KC)              // 8 chunks

__global__ void __launch_bounds__(256)
k_logits_gemm(const __bf16* __restrict__ hsB,    // [B*T, HID]
              const __bf16* __restrict__ WoutB,  // [VOCAB, HID]
              const float*  __restrict__ b_out,  // [VOCAB]
              float*        __restrict__ out) {  // [B*T, VOCAB]
  __shared__ __align__(16) __bf16 btile[2][80][KC];   // 40 KB

  const int lane = threadIdx.x & 31;
  const int wid  = threadIdx.x >> 5;
  const int row0 = (blockIdx.y * 8 + wid) * 16;
  const int n0   = blockIdx.x * 80;

  const __bf16* aPtr = hsB + (size_t)(row0 + (lane & 15)) * HID;

  f32x8 acc[5];
  f32x8 z = {};
#pragma unroll
  for (int j = 0; j < 5; ++j) acc[j] = z;

  // prime the pipeline: TDM chunk 0 -> buffer 0
  if (wid == 0) {
    tdm_load_2d((unsigned)(size_t)&btile[0][0][0],
                WoutB + (size_t)n0 * HID, KC, 80, HID, HID, VOCAB);
  }

  for (int c = 0; c < NKC; ++c) {
    const int buf = c & 1;
    if (wid == 0) __builtin_amdgcn_s_wait_tensorcnt(0);
    __syncthreads();                       // chunk c visible in btile[buf]

    if (wid == 0 && (c + 1) < NKC) {       // overlap: fetch chunk c+1
      tdm_load_2d((unsigned)(size_t)&btile[buf ^ 1][0][0],
                  WoutB + (size_t)n0 * HID + (size_t)(c + 1) * KC,
                  KC, 80, HID, HID, VOCAB);
    }

    const int kbase = c * KC;
#pragma unroll
    for (int kt = 0; kt < KC / 32; ++kt) {
      bf16x16 a = load_frag_row(aPtr, kbase + kt * 32, lane);
#pragma unroll
      for (int j = 0; j < 5; ++j) {
        const __bf16* brow = &btile[buf][j * 16 + (lane & 15)][0];
        bf16x16 b = load_frag_row(brow, kt * 32, lane);
        acc[j] = wmma_bf16(a, b, acc[j]);
      }
    }
    __syncthreads();                       // all waves done with btile[buf]
  }

  const int mAdd = (lane & 16) ? 8 : 0;
#pragma unroll
  for (int j = 0; j < 5; ++j) {
    int col = n0 + j * 16 + (lane & 15);
    float bias = b_out[col];
#pragma unroll
    for (int r = 0; r < 8; ++r) {
      int row = row0 + r + mAdd;
      out[(size_t)row * VOCAB + col] = acc[j][r] + bias;
    }
  }
}

// ---------------------------------------------------------------------------
// Host launcher
// ---------------------------------------------------------------------------
extern "C" void kernel_launch(void* const* d_in, const int* in_sizes, int n_in,
                              void* d_out, int out_size, void* d_ws, size_t ws_size,
                              hipStream_t stream) {
  const int*   x     = (const int*)  d_in[0];   // [B, T]
  const float* h0    = (const float*)d_in[1];   // [B, HID]
  const float* emb   = (const float*)d_in[2];   // [VOCAB, EMB]
  const float* W_ih  = (const float*)d_in[3];   // [HID, EMB]
  const float* b_ih  = (const float*)d_in[4];   // [HID]
  const float* W_hh  = (const float*)d_in[5];   // [HID, HID]
  const float* b_hh  = (const float*)d_in[6];   // [HID]
  const float* W_out = (const float*)d_in[7];   // [VOCAB, HID]
  const float* b_out = (const float*)d_in[8];   // [VOCAB]

  float* logits = (float*)d_out;                           // [B*T, VOCAB]
  float* hT     = (float*)d_out + (size_t)MROWS * VOCAB;   // [B, HID]

  char* ws = (char*)d_ws;
  size_t off = 0;
  auto carve = [&](size_t bytes) {
    char* p = ws + off;
    off += (bytes + 255) & ~(size_t)255;
    return p;
  };
  __bf16* embB  = (__bf16*)carve((size_t)VOCAB * EMB * 2);
  __bf16* WihB  = (__bf16*)carve((size_t)HID * EMB * 2);
  __bf16* WhhB  = (__bf16*)carve((size_t)HID * HID * 2);
  __bf16* WoutB = (__bf16*)carve((size_t)VOCAB * HID * 2);
  float*  xiF   = (float*) carve((size_t)MROWS * HID * 4);
  __bf16* hsB   = (__bf16*)carve((size_t)MROWS * HID * 2);
  (void)ws_size; (void)in_sizes; (void)n_in; (void)out_size;

  // 0) convert weights / embeddings to bf16
  k_f32_to_bf16<<<2048, 256, 0, stream>>>(emb,   embB,  VOCAB * EMB);
  k_f32_to_bf16<<<1024, 256, 0, stream>>>(W_ih,  WihB,  HID * EMB);
  k_f32_to_bf16<<<1024, 256, 0, stream>>>(W_hh,  WhhB,  HID * HID);
  k_f32_to_bf16<<<2048, 256, 0, stream>>>(W_out, WoutB, VOCAB * HID);

  // 1) input projection
  k_xi_gemm<<<dim3(MROWS / 128, HID / 64), 256, 0, stream>>>(x, embB, WihB, b_ih, xiF);

  // 2) sequential recurrence, single WGP
  k_recurrence<<<1, 1024, 0, stream>>>(xiF, h0, WhhB, b_hh, hsB, hT);

  // 3) output projection (TDM-staged B tiles)
  k_logits_gemm<<<dim3(VOCAB / 80, MROWS / 128), 256, 0, stream>>>(hsB, WoutB, b_out, logits);
}
